// AttentionHead_22076131901760
// MI455X (gfx1250) — matrix operand order repaired and checked by reading, not compiled, not run on hardware
//
#include <hip/hip_runtime.h>
#include <hip/hip_bf16.h>

#define NB 4
#define SEQ 4096
#define DM 1024
#define DK 128

static constexpr float kScale = 0.08838834764831843f;  // 1/sqrt(128)

typedef __attribute__((ext_vector_type(16))) __bf16 v16bf_t;
typedef __attribute__((ext_vector_type(8)))  __bf16 v8bf_t;
typedef __attribute__((ext_vector_type(8)))  float  v8f_t;
typedef __attribute__((ext_vector_type(4)))  unsigned int u32x4;

union Frag32 { v16bf_t v; u32x4 u[2]; };
union Pack16 { v8bf_t  v; u32x4 u; };

__device__ __forceinline__ v8f_t wmma_bf16(v16bf_t a, v16bf_t b, v8f_t c) {
  return __builtin_amdgcn_wmma_f32_16x16x32_bf16(false, a, false, b, (short)0, c,
                                                 false, false);
}

// ---------------- kernel 0: repack weights ----------------
// wt[p][n][k] = W_p[k][n] as bf16, p in {q,k,v}; makes B-fragment loads contiguous.
__global__ __launch_bounds__(256) void prep_weights(
    const float* __restrict__ Wq, const float* __restrict__ Wk,
    const float* __restrict__ Wv, __bf16* __restrict__ wt) {
  int idx = blockIdx.x * 256 + threadIdx.x;
  if (idx >= 3 * DK * DM) return;
  int p = idx / (DK * DM);
  int rem = idx - p * (DK * DM);
  int n = rem / DM;
  int k = rem - n * DM;
  const float* W = (p == 0) ? Wq : (p == 1) ? Wk : Wv;
  wt[idx] = (__bf16)W[k * DK + n];
}

// ---------------- kernel 1: QKV projection ----------------
// block = 16 rows of x (staged once to LDS as bf16); 8 waves, wave w owns
// d-column tile w of Q, K and V (three 16x16 accumulators, K-loop of 32).
// outputs: qbuf [NB][SEQ][DK] bf16 (Q*scale, row-major)
//          kbuf [NB][SEQ][DK] bf16 (row-major)
//          vtb  [NB][DK][SEQ] bf16 (transposed)
__global__ __launch_bounds__(256) void qkv_proj(
    const float* __restrict__ x, const __bf16* __restrict__ wt,
    const float* __restrict__ bq, const float* __restrict__ bk,
    const float* __restrict__ bv, __bf16* __restrict__ qbuf,
    __bf16* __restrict__ kbuf, __bf16* __restrict__ vtb) {
  __shared__ __bf16 xs[16 * DM];  // 32 KB of the 320 KB WGP LDS
  const int tid = threadIdx.x;
  const int row0 = blockIdx.x * 16;  // flat row in [0, NB*SEQ)

  {  // stage x tile -> LDS (f32 -> bf16), each thread 64 consecutive floats
    int r = tid >> 4;
    int ch = tid & 15;
    const float* src = x + (long)(row0 + r) * DM + ch * 64;
    __bf16* dst = xs + r * DM + ch * 64;
#pragma unroll
    for (int i = 0; i < 64; i += 4) {
      float4 f = *(const float4*)(src + i);
      dst[i + 0] = (__bf16)f.x;
      dst[i + 1] = (__bf16)f.y;
      dst[i + 2] = (__bf16)f.z;
      dst[i + 3] = (__bf16)f.w;
    }
  }
  __syncthreads();

  const int wave = tid >> 5, lane = tid & 31;
  const int ln = lane & 15, lh = lane >> 4;
  const int n0 = wave * 16;

  const __bf16* wq = wt;
  const __bf16* wk = wt + DK * DM;
  const __bf16* wv = wt + 2 * DK * DM;

  v8f_t cq = {}, ck = {}, cv = {};
  for (int c = 0; c < DM; c += 32) {
    Frag32 a, fq, fk, fv;
    // A frag (16x32 bf16): lane<16 holds K {0..15}, lane>=16 {16..31}
    const __bf16* ap = xs + ln * DM + c + 16 * lh;
    a.u[0] = *(const u32x4*)ap;
    a.u[1] = *(const u32x4*)(ap + 8);
    // B frag (32x16): lane ln = column, 16 contiguous K values at c + 16*lh
    const int woff = (n0 + ln) * DM + c + 16 * lh;
    fq.u[0] = *(const u32x4*)(wq + woff);
    fq.u[1] = *(const u32x4*)(wq + woff + 8);
    fk.u[0] = *(const u32x4*)(wk + woff);
    fk.u[1] = *(const u32x4*)(wk + woff + 8);
    fv.u[0] = *(const u32x4*)(wv + woff);
    fv.u[1] = *(const u32x4*)(wv + woff + 8);
    cq = wmma_bf16(a.v, fq.v, cq);
    ck = wmma_bf16(a.v, fk.v, ck);
    cv = wmma_bf16(a.v, fv.v, cv);
  }

  const int b = row0 / SEQ;
  const int s0 = row0 % SEQ;
  const float bqv = bq[n0 + ln], bkv = bk[n0 + ln], bvv = bv[n0 + ln];

  // Q (pre-scaled) and K: row-major scatter stores (C layout: row = r + 8*lh)
#pragma unroll
  for (int r = 0; r < 8; r++) {
    long base = ((long)b * SEQ + s0 + r + 8 * lh) * DK + n0 + ln;
    qbuf[base] = (__bf16)((cq[r] + bqv) * kScale);
    kbuf[base] = (__bf16)(ck[r] + bkv);
  }
  // V transposed: the 8 accumulator rows are contiguous s -> one 16B store
  Pack16 pv;
#pragma unroll
  for (int r = 0; r < 8; r++) pv.v[r] = (__bf16)(cv[r] + bvv);
  *(u32x4*)(vtb + ((long)b * DK + n0 + ln) * SEQ + s0 + 8 * lh) = pv.u;
}

// ---------------- kernel 2: flash attention ----------------
// one wave = one 16-row q tile; k processed in blocks of 32 (causal-trimmed).
__global__ __launch_bounds__(128) void attn_fwd(
    const __bf16* __restrict__ qbuf, const __bf16* __restrict__ kbuf,
    const __bf16* __restrict__ vtb, const float* __restrict__ pm,
    float* __restrict__ out) {
  __shared__ __bf16 pbuf[4][16 * 32];  // wave-private P staging, 1 KB each
  const int tid = threadIdx.x;
  const int wave = tid >> 5, lane = tid & 31;
  const int ln = lane & 15, lh = lane >> 4;
  const int g = blockIdx.x * 4 + wave;  // q-tile id, 0..1023
  const int b = g >> 8;                 // SEQ/16 = 256 tiles per batch
  const int q0 = (g & 255) << 4;

  // Q fragments resident for the whole kernel (row q0+ln, all 128 d)
  Frag32 qa[4];
  const __bf16* qrow = qbuf + ((long)b * SEQ + q0 + ln) * DK;
#pragma unroll
  for (int c = 0; c < 4; c++) {
    const __bf16* ap = qrow + c * 32 + 16 * lh;
    qa[c].u[0] = *(const u32x4*)ap;
    qa[c].u[1] = *(const u32x4*)(ap + 8);
  }

  float pmq[8];
#pragma unroll
  for (int r = 0; r < 8; r++) pmq[r] = pm[b * SEQ + q0 + r + 8 * lh];

  float m_i[8], l_i[8];
  v8f_t O[8];
  v8f_t vzero = {};
#pragma unroll
  for (int r = 0; r < 8; r++) { m_i[r] = -1e20f; l_i[r] = 0.f; }
#pragma unroll
  for (int dt = 0; dt < 8; dt++) O[dt] = vzero;

  const __bf16* kbb = kbuf + (long)b * SEQ * DK;
  const __bf16* vbb = vtb + (long)b * DK * SEQ;
  const float* pmk = pm + b * SEQ;
  __bf16* pw = pbuf[wave];

  const int kend = q0 + 16;  // blocks with kblk <= q0+15
  for (int kblk = 0; kblk < kend; kblk += 32) {
    if (kblk + 32 < kend)
      __builtin_prefetch(kbb + (long)(kblk + 32 + ln) * DK, 0, 0);

    // scores tile pair [16q x 16k] x2 : S = Q * K^T, contraction over d
    v8f_t t0 = vzero, t1 = vzero;
#pragma unroll
    for (int c = 0; c < 4; c++) {
      Frag32 f0, f1;  // B frags: column = k index, 16 contiguous d values
      const __bf16* k0p = kbb + (long)(kblk + ln) * DK + c * 32 + 16 * lh;
      const __bf16* k1p = k0p + 16 * DK;
      f0.u[0] = *(const u32x4*)k0p; f0.u[1] = *(const u32x4*)(k0p + 8);
      f1.u[0] = *(const u32x4*)k1p; f1.u[1] = *(const u32x4*)(k1p + 8);
      t0 = wmma_bf16(qa[c].v, f0.v, t0);
      t1 = wmma_bf16(qa[c].v, f1.v, t1);
    }

    // causal + padding masks (scale already folded into Q)
    const float pk0 = pmk[kblk + ln];
    const float pk1 = pmk[kblk + 16 + ln];
    float bmax[8];
#pragma unroll
    for (int r = 0; r < 8; r++) {
      const int q = q0 + r + 8 * lh;
      const bool rowok = (pmq[r] > 0.5f);
      const bool a0 = rowok && (pk0 > 0.5f) && (kblk + ln <= q);
      const bool a1 = rowok && (pk1 > 0.5f) && (kblk + 16 + ln <= q);
      t0[r] = a0 ? t0[r] : -1e30f;
      t1[r] = a1 ? t1[r] : -1e30f;
      bmax[r] = fmaxf(t0[r], t1[r]);
    }
    // row max: rows live within one 16-lane half -> 4 xor-shuffle steps
#pragma unroll
    for (int off = 1; off < 16; off <<= 1)
#pragma unroll
      for (int r = 0; r < 8; r++)
        bmax[r] = fmaxf(bmax[r], __shfl_xor(bmax[r], off, 32));

    float corr[8], rsum[8];
#pragma unroll
    for (int r = 0; r < 8; r++) {
      float mn = fmaxf(m_i[r], bmax[r]);  // m_i >= -1e20 keeps masked exp -> 0
      corr[r] = __expf(m_i[r] - mn);
      m_i[r] = mn;
      t0[r] = __expf(t0[r] - mn);
      t1[r] = __expf(t1[r] - mn);
      rsum[r] = t0[r] + t1[r];
    }
#pragma unroll
    for (int off = 1; off < 16; off <<= 1)
#pragma unroll
      for (int r = 0; r < 8; r++)
        rsum[r] += __shfl_xor(rsum[r], off, 32);
#pragma unroll
    for (int r = 0; r < 8; r++) l_i[r] = l_i[r] * corr[r] + rsum[r];

    // P: C-layout -> row-major [16q x 32k] bf16 in LDS, read back as A frag
#pragma unroll
    for (int r = 0; r < 8; r++) {
      pw[(r + 8 * lh) * 32 + ln] = (__bf16)t0[r];
      pw[(r + 8 * lh) * 32 + 16 + ln] = (__bf16)t1[r];
    }
    __asm__ __volatile__("" ::: "memory");  // keep DS store->load order in IR
    Frag32 pa;
    const __bf16* pr = pw + ln * 32 + 16 * lh;
    pa.u[0] = *(const u32x4*)pr;
    pa.u[1] = *(const u32x4*)(pr + 8);

    // O[dt] = O[dt]*corr + P x V ; V B-frags contiguous from transposed vtb
#pragma unroll
    for (int dt = 0; dt < 8; dt++) {
      Frag32 vf;
      const __bf16* vp = vbb + (long)(dt * 16 + ln) * SEQ + kblk + 16 * lh;
      vf.u[0] = *(const u32x4*)vp;
      vf.u[1] = *(const u32x4*)(vp + 8);
      v8f_t o = O[dt];
#pragma unroll
      for (int r = 0; r < 8; r++) o[r] *= corr[r];
      O[dt] = wmma_bf16(pa.v, vf.v, o);
    }
  }

  // epilogue: normalize; dead (padded) rows have l==0 -> emit 0 like reference
#pragma unroll
  for (int dt = 0; dt < 8; dt++) {
#pragma unroll
    for (int r = 0; r < 8; r++) {
      const int q = q0 + r + 8 * lh;
      const float val = (l_i[r] > 0.f) ? O[dt][r] / l_i[r] : 0.f;
      out[((long)b * SEQ + q) * DK + dt * 16 + ln] = val;
    }
  }
}

extern "C" void kernel_launch(void* const* d_in, const int* in_sizes, int n_in,
                              void* d_out, int out_size, void* d_ws, size_t ws_size,
                              hipStream_t stream) {
  const float* x  = (const float*)d_in[0];
  const float* pm = (const float*)d_in[1];
  const float* Wq = (const float*)d_in[2];
  const float* bq = (const float*)d_in[3];
  const float* Wk = (const float*)d_in[4];
  const float* bk = (const float*)d_in[5];
  const float* Wv = (const float*)d_in[6];
  const float* bv = (const float*)d_in[7];
  float* out = (float*)d_out;

  char* ws = (char*)d_ws;
  __bf16* wt   = (__bf16*)(ws);               // 768 KB: repacked weights
  __bf16* qbuf = (__bf16*)(ws + (1u << 20));  // 4 MB: Q bf16 row-major (scaled)
  __bf16* kbuf = (__bf16*)(ws + (5u << 20));  // 4 MB: K bf16 row-major
  __bf16* vtb  = (__bf16*)(ws + (9u << 20));  // 4 MB: V bf16 transposed [d][s]

  prep_weights<<<(3 * DK * DM + 255) / 256, 256, 0, stream>>>(Wq, Wk, Wv, wt);
  qkv_proj<<<NB * SEQ / 16, 256, 0, stream>>>(x, wt, bq, bk, bv, qbuf, kbuf, vtb);
  attn_fwd<<<NB * (SEQ / 16) / 4, 128, 0, stream>>>(qbuf, kbuf, vtb, pm, out);

  (void)in_sizes; (void)n_in; (void)out_size; (void)ws_size;
}
